// Swin_FDWA_1812476199059
// MI455X (gfx1250) — matrix operand order repaired and checked by reading, not compiled, not run on hardware
//
#include <hip/hip_runtime.h>
#include <hip/hip_bf16.h>
#include <math.h>

typedef __attribute__((ext_vector_type(16))) __bf16 v16bf;
typedef __attribute__((ext_vector_type(8)))  float  v8f;

union FragBF { v16bf v; __bf16 e[16]; uint4 q[2]; };

// CDNA5 async global->LDS copy (ASYNCcnt-tracked), feature-gated.
#if defined(__AMDGCN__) && __has_builtin(__builtin_amdgcn_global_load_async_to_lds_b128)
#define HAVE_ASYNC_LDS 1
#else
#define HAVE_ASYNC_LDS 0
#endif

// Builtin's exact pointee type (from hipcc diagnostic):
//   __attribute__((__vector_size__(4 * sizeof(int)))) int
typedef int v4i_gnu __attribute__((vector_size(16)));

__device__ __forceinline__ __bf16 f2bf(float f){
  unsigned u = __builtin_bit_cast(unsigned, f);
  unsigned r = (u + 0x7FFFu + ((u >> 16) & 1u)) >> 16;
  unsigned short h = (unsigned short)r;
  return __builtin_bit_cast(__bf16, h);
}

__device__ __forceinline__ v8f v8f_zero(){
  v8f z;
#pragma unroll
  for (int i = 0; i < 8; ++i) z[i] = 0.0f;
  return z;
}

__device__ __forceinline__ v8f wmma_bf16(v16bf a, v16bf b, v8f c){
  // v_wmma_f32_16x16x32_bf16: D = A(16x32) * B(32x16) + C(16x16 f32)
  return __builtin_amdgcn_wmma_f32_16x16x32_bf16(false, a, false, b, (short)0, c, false, false);
}

// A fragment (16-bit A 16x32): lanes 0-15 row M=lane, K k0..k0+15;
// lanes 16-31 row M=lane-16, K k0+16..k0+31. Row-major A[M][lda].
__device__ __forceinline__ v16bf load_a_frag(const __bf16* A, int lda, int m0, int k0){
  int lane = (int)threadIdx.x & 31;
  int row  = m0 + (lane & 15);
  int kb   = k0 + ((lane & 16) ? 16 : 0);
  const __bf16* p = A + (size_t)row * lda + kb;
  FragBF f;
  f.q[0] = *(const uint4*)(p);
  f.q[1] = *(const uint4*)(p + 8);
  return f.v;
}

// B fragment (32x16, 16-bit): lanes 0-15 col N=lane, K k0..k0+15; lanes 16-31 col N=lane-16,
// K k0+16..k0+31. Weights stored row-major W[N][ldw] (out = x @ W^T), so loads are contiguous.
__device__ __forceinline__ v16bf load_b_frag(const __bf16* W, int ldw, int n0, int k0){
  int lane = (int)threadIdx.x & 31;
  int col  = n0 + (lane & 15);
  int kb   = k0 + ((lane & 16) ? 16 : 0);
  const __bf16* p = W + (size_t)col * ldw + kb;
  FragBF f;
  f.q[0] = *(const uint4*)(p);
  f.q[1] = *(const uint4*)(p + 8);
  return f.v;
}

// B fragment out of an LDS tile stored [64 cols][32 K] bf16.
__device__ __forceinline__ v16bf load_b_lds(const __bf16* base, int j){
  int lane = (int)threadIdx.x & 31;
  int col  = (j << 4) + (lane & 15);
  int kb   = (lane & 16) ? 16 : 0;
  const __bf16* p = base + col * 32 + kb;
  FragBF f;
  f.q[0] = *(const uint4*)(p);
  f.q[1] = *(const uint4*)(p + 8);
  return f.v;
}

// One 16-byte slice of the block's 64x32 weight tile, staged to LDS.
__device__ __forceinline__ void stage_b16(const __bf16* g, __bf16* l){
#if HAVE_ASYNC_LDS
  // signature (probe-derived): (global v4i*, lds v4i*, imm offset, imm cpol)
  __builtin_amdgcn_global_load_async_to_lds_b128(
      (__attribute__((address_space(1))) v4i_gnu*)g,
      (__attribute__((address_space(3))) v4i_gnu*)l, 0, 0);
#else
  *(uint4*)l = *(const uint4*)g;
#endif
}

template<int N>
__device__ __forceinline__ void wait_async(){
#if HAVE_ASYNC_LDS
# if __has_builtin(__builtin_amdgcn_s_wait_asynccnt)
  __builtin_amdgcn_s_wait_asynccnt(N);
# else
  asm volatile("s_wait_asynccnt %0" :: "i"(N) : "memory");
# endif
#endif
}

// ---------------------------------------------------------------------------
// bf16 WMMA GEMM: out[M,N] = A[M,K] @ W[N,K]^T (+ epilogue)
// Block tile 256(M) x 64(N): 8 waves, each 32x64 (8 WMMAs per 32-K step).
// Weight tile (64x32 = 4 KB) staged once per block into a double-buffered LDS
// tile via CDNA5 async global->LDS copies, overlapped with WMMA of the
// previous K-step. A operands stream straight from global (b128 clauses).
// ---------------------------------------------------------------------------
enum { EPI_STORE_BF16 = 0, EPI_BIAS_RES_F32 = 1, EPI_BIAS_GELU_BF16 = 2, EPI_BIAS_F32 = 3 };

template<int EPI>
__global__ __launch_bounds__(256) void gemm_wmma(
    const __bf16* __restrict__ A, const __bf16* __restrict__ W,
    const float* __restrict__ bias, const float* __restrict__ res,
    void* __restrict__ outp, int M, int N, int K){
  __shared__ __bf16 btile[2][64 * 32];   // [parity][col*32 + k]

  int tid  = (int)threadIdx.x;
  int wvb  = tid >> 5;
  int lane = tid & 31;
  int nt   = N >> 6;
  int bm   = (int)blockIdx.x / nt;
  int bn   = (int)blockIdx.x - bm * nt;
  int m0   = bm * 256 + wvb * 32;
  int n0   = bn * 64;

  // this thread's 16B slice of the 64x32 weight tile
  const __bf16* gsl = W + (size_t)(n0 + (tid >> 2)) * K + ((tid & 3) << 3);
  __bf16* lsl[2] = { &btile[0][tid * 8], &btile[1][tid * 8] };

  v8f acc[2][4];
#pragma unroll
  for (int h = 0; h < 2; ++h)
#pragma unroll
    for (int j = 0; j < 4; ++j) acc[h][j] = v8f_zero();

  stage_b16(gsl, lsl[0]);               // preload K-step 0

  for (int k0 = 0; k0 < K; k0 += 32){
    int par = (k0 >> 5) & 1;
    if (k0 + 32 < K){                   // overlap copy of next tile with math
      stage_b16(gsl + k0 + 32, lsl[par ^ 1]);
      wait_async<1>();
    } else {
      wait_async<0>();
    }
    __syncthreads();                    // all waves' slices of tile k0 visible

    if (k0 + 32 < K)                    // gfx1250 global_prefetch_b8
      __builtin_prefetch(A + (size_t)(m0 + lane) * K + k0 + 32, 0, 1);
    v16bf a0 = load_a_frag(A, K, m0, k0);
    v16bf a1 = load_a_frag(A, K, m0 + 16, k0);
    const __bf16* bt = &btile[par][0];
#pragma unroll
    for (int j = 0; j < 4; ++j){
      v16bf b = load_b_lds(bt, j);
      acc[0][j] = wmma_bf16(a0, b, acc[0][j]);
      acc[1][j] = wmma_bf16(a1, b, acc[1][j]);
    }
    __syncthreads();                    // reads done before tile is overwritten
  }

  int rofs = (lane & 16) ? 8 : 0;       // C layout: lanes16-31 hold rows 8..15
  int cb   = n0 + (lane & 15);
#pragma unroll
  for (int h = 0; h < 2; ++h){
    int mbase = m0 + (h << 4);
#pragma unroll
    for (int j = 0; j < 4; ++j){
      int col = cb + (j << 4);
      float bv = (EPI != EPI_STORE_BF16) ? bias[col] : 0.0f;
#pragma unroll
      for (int r = 0; r < 8; ++r){
        int row = mbase + rofs + r;
        float v = acc[h][j][r] + bv;
        size_t idx = (size_t)row * N + col;
        if (EPI == EPI_STORE_BF16){
          ((__bf16*)outp)[idx] = f2bf(v);
        } else if (EPI == EPI_BIAS_RES_F32){
          ((float*)outp)[idx] = v + res[idx];
        } else if (EPI == EPI_BIAS_GELU_BF16){
          float g = 0.5f * v * (1.0f + erff(v * 0.70710678118654752f)); // exact GELU
          ((__bf16*)outp)[idx] = f2bf(g);
        } else {
          ((float*)outp)[idx] = v;
        }
      }
    }
  }
}

// ---------------------------------------------------------------------------
// LayerNorm over C=256, one row per 256-thread block, output bf16.
// ---------------------------------------------------------------------------
__global__ __launch_bounds__(256) void layernorm_bf16(
    const float* __restrict__ x, const float* __restrict__ g,
    const float* __restrict__ bta, __bf16* __restrict__ out){
  __shared__ float ps[8], pq[8];
  int row = blockIdx.x, tid = threadIdx.x;
  float v = x[(size_t)row * 256 + tid];
  float s = v, q = v * v;
#pragma unroll
  for (int d = 1; d < 32; d <<= 1){ s += __shfl_xor(s, d, 32); q += __shfl_xor(q, d, 32); }
  if ((tid & 31) == 0){ ps[tid >> 5] = s; pq[tid >> 5] = q; }
  __syncthreads();
  float st = 0.f, qt = 0.f;
#pragma unroll
  for (int i = 0; i < 8; ++i){ st += ps[i]; qt += pq[i]; }
  float mean = st * (1.0f / 256.0f);
  float var  = qt * (1.0f / 256.0f) - mean * mean;
  float r = rsqrtf(var + 1e-5f);
  out[(size_t)row * 256 + tid] = f2bf((v - mean) * r * g[tid] + bta[tid]);
}

// f32 -> bf16 weight conversion
__global__ void cvt_f32_bf16(const float* __restrict__ in, __bf16* __restrict__ out, int n){
  int i = blockIdx.x * 256 + threadIdx.x;
  if (i < n) out[i] = f2bf(in[i]);
}

// Expand bias table -> dense bias[head][i][j] using the rel_index formula.
__global__ void build_bias(const float* __restrict__ bt, float* __restrict__ out, int Hs, int Ws){
  int S = Hs * Ws;
  int idx = blockIdx.x * 256 + threadIdx.x;
  if (idx >= 2 * S * S) return;
  int head = idx / (S * S);
  int r = idx - head * S * S;
  int i = r / S, j = r - (r / S) * S;
  int dh = (i / Ws - j / Ws) + Hs - 1;
  int dw = (i % Ws) - (j % Ws) + Ws - 1;
  int rel = dh * (2 * Ws - 1) + dw;
  out[idx] = bt[rel * 2 + head];
}

// ---------------------------------------------------------------------------
// Shifted-window attention, all-WMMA. One wave per (window, head, 16-row
// query tile). hd=32 -> one WMMA K-step for q@k^T; scores kept in registers;
// softmax row-reductions via shfl_xor within the 16-lane halves of the C
// layout; P staged through LDS (bf16) to convert C-layout -> A-layout for
// the P@V WMMAs. SHH/SHW are the positive source shifts (= -SHIFTS[i]).
// qkv layout: bf16 [M=B*L][3*256], s-chunk at +s*256.
// ---------------------------------------------------------------------------
template<int HS, int WS, int SHH, int SHW>
__global__ __launch_bounds__(128) void win_attn_wmma(
    const __bf16* __restrict__ qkv, const float* __restrict__ bias,
    __bf16* __restrict__ att, int cbase){
  constexpr int S   = HS * WS;           // tokens per window
  constexpr int SN  = S / 16;            // key tiles
  constexpr int QT  = S / 16;            // query tiles
  constexpr int SP  = (S < 32) ? 32 : S; // padded key count for K=32 WMMA
  constexpr int NWH = 192 / HS, NWW = 192 / WS;
  __shared__ __bf16 plds[4][16][SP];

  int wvb  = (int)threadIdx.x >> 5;
  int wv   = (blockIdx.x << 2) + wvb;
  int lane = (int)threadIdx.x & 31;
  int win  = wv / (2 * QT);
  int rem  = wv - win * (2 * QT);
  int head = rem / QT;
  int qt   = rem - head * QT;
  int bb   = win / (NWH * NWW);
  int wr   = win - bb * (NWH * NWW);
  int wh0  = (wr / NWW) * HS;
  int ww0  = (wr - (wr / NWW) * NWW) * WS;
  int ch   = cbase + head * 32;          // head channel base within 256

  auto tok = [&](int t) -> int {         // window token -> global pixel index
    int th = t / WS, tw = t - th * WS;
    int h = wh0 + th + SHH; if (h >= 192) h -= 192;
    int w = ww0 + tw + SHW; if (w >= 192) w -= 192;
    return bb * (192 * 192) + h * 192 + w;
  };

  int jc   = lane & 15;
  int rofs = (lane & 16) ? 8 : 0;

  // Q fragment (A layout), row = query token
  FragBF qa;
  {
    int tq = qt * 16 + jc;
    const __bf16* p = qkv + (size_t)tok(tq) * 768 + ch + ((lane & 16) ? 16 : 0);
    qa.q[0] = *(const uint4*)(p);
    qa.q[1] = *(const uint4*)(p + 8);
  }

  // scores S = q @ k^T  (one WMMA per 16-key tile, K = hd = 32)
  v8f sc[SN];
#pragma unroll
  for (int kt = 0; kt < SN; ++kt){
    FragBF kf;
    int tk = kt * 16 + jc;
    const __bf16* p = qkv + (size_t)tok(tk) * 768 + 256 + ch + ((lane & 16) ? 16 : 0);
    kf.q[0] = *(const uint4*)(p);
    kf.q[1] = *(const uint4*)(p + 8);
    sc[kt] = wmma_bf16(qa.v, kf.v, v8f_zero());
  }

  // scale + relative-position bias
  const float scale = 0.17677669529663687f; // hd^-0.5, hd = 32
  const float* bh = bias + (size_t)head * S * S;
#pragma unroll
  for (int kt = 0; kt < SN; ++kt){
#pragma unroll
    for (int r = 0; r < 8; ++r){
      int i = qt * 16 + rofs + r;
      int j = kt * 16 + jc;
      sc[kt][r] = sc[kt][r] * scale + bh[i * S + j];
    }
  }

  // softmax over keys; row = (r, lane half); reduce across the 16-lane half
  float lr[8];
#pragma unroll
  for (int r = 0; r < 8; ++r){
    float m = -1e30f;
#pragma unroll
    for (int kt = 0; kt < SN; ++kt) m = fmaxf(m, sc[kt][r]);
#pragma unroll
    for (int d = 1; d < 16; d <<= 1) m = fmaxf(m, __shfl_xor(m, d, 32));
    float ssum = 0.f;
#pragma unroll
    for (int kt = 0; kt < SN; ++kt){
      float e = __expf(sc[kt][r] - m);
      sc[kt][r] = e;
      ssum += e;
    }
#pragma unroll
    for (int d = 1; d < 16; d <<= 1) ssum += __shfl_xor(ssum, d, 32);
    lr[r] = ssum;
  }

  // stage P (bf16) in LDS, row-major [16 q-rows][SP keys]
#pragma unroll
  for (int kt = 0; kt < SN; ++kt)
#pragma unroll
    for (int r = 0; r < 8; ++r)
      plds[wvb][rofs + r][kt * 16 + jc] = f2bf(sc[kt][r]);
  if constexpr (S < 32){               // zero-pad keys S..31
#pragma unroll
    for (int r = 0; r < 8; ++r)
      plds[wvb][rofs + r][16 + jc] = f2bf(0.0f);
  }
  __syncthreads();

  // O = P @ V  (two 16-col output tiles since hd = 32)
  v8f o0 = v8f_zero(), o1 = v8f_zero();
#pragma unroll
  for (int k2 = 0; k2 < SP; k2 += 32){
    FragBF pa;
    {
      const __bf16* p = &plds[wvb][jc][k2 + ((lane & 16) ? 16 : 0)];
      pa.q[0] = *(const uint4*)(p);
      pa.q[1] = *(const uint4*)(p + 8);
    }
    int kbase = k2 + ((lane & 16) ? 16 : 0);
    FragBF v0f, v1f;
#pragma unroll
    for (int kk = 0; kk < 16; ++kk){
      int key = kbase + kk;
      if (S >= 32 || key < S){
        const __bf16* vp = qkv + (size_t)tok(key) * 768 + 512 + ch;
        v0f.e[kk] = vp[jc];
        v1f.e[kk] = vp[16 + jc];
      } else {
        v0f.e[kk] = f2bf(0.0f);
        v1f.e[kk] = f2bf(0.0f);
      }
    }
    o0 = wmma_bf16(pa.v, v0f.v, o0);
    o1 = wmma_bf16(pa.v, v1f.v, o1);
  }

  // normalize by row sum and scatter back through the inverse roll
#pragma unroll
  for (int r = 0; r < 8; ++r){
    int trow = qt * 16 + rofs + r;
    size_t base = (size_t)tok(trow) * 256 + cbase + head * 32;
    float inv = 1.0f / lr[r];
    att[base + jc]      = f2bf(o0[r] * inv);
    att[base + 16 + jc] = f2bf(o1[r] * inv);
  }
}

// ---------------------------------------------------------------------------
// freq_mod: per 16x16 patch, per channel: irfft2(rfft2(x)*w) (ortho), + x1.
// Block = 256 threads = 16 channels x 16 rows for one patch; two LDS-staged
// separable DFT passes. Writes every element of d_out.
// ---------------------------------------------------------------------------
__global__ __launch_bounds__(256) void freq_mod_add(
    const float* __restrict__ h2, const float* __restrict__ fmw,
    const float* __restrict__ x1, float* __restrict__ out){
  __shared__ float xs[16][16][16];      // [c][h][w]
  __shared__ float fs[16][16][9][2];    // [c][kh][kw][re/im]
  int patch = blockIdx.x;               // 0..287 (B * 12 * 12)
  int cg = blockIdx.y;                  // 0..15
  int b  = patch / 144;
  int pr = patch - b * 144;
  int ph = pr / 12, pw = pr - (pr / 12) * 12;
  int ci = threadIdx.x & 15;
  int c  = cg * 16 + ci;
  int t  = (int)threadIdx.x >> 4;
  const float w16 = 0.39269908169872414f; // 2*pi/16

  { // load one spatial row (h = t) of this channel
    size_t rb = ((size_t)(b * 192 + ph * 16 + t) * 192 + pw * 16) * 256 + c;
#pragma unroll
    for (int w = 0; w < 16; ++w) xs[ci][t][w] = h2[rb + (size_t)w * 256];
  }
  __syncthreads();

  { // forward: DFT over h (kh = t), rDFT over w, multiply complex weight
    int kh = t;
    float f1r[16], f1i[16];
#pragma unroll
    for (int w = 0; w < 16; ++w){ f1r[w] = 0.f; f1i[w] = 0.f; }
    for (int h = 0; h < 16; ++h){
      float a = -w16 * (float)((kh * h) & 15);
      float cs = __cosf(a), sn = __sinf(a);
#pragma unroll
      for (int w = 0; w < 16; ++w){
        float xv = xs[ci][h][w];
        f1r[w] += xv * cs;
        f1i[w] += xv * sn;
      }
    }
#pragma unroll
    for (int kw = 0; kw < 9; ++kw){
      float Fr = 0.f, Fi = 0.f;
      for (int w = 0; w < 16; ++w){
        float a = -w16 * (float)((kw * w) & 15);
        float cs = __cosf(a), sn = __sinf(a);
        Fr += f1r[w] * cs - f1i[w] * sn;
        Fi += f1r[w] * sn + f1i[w] * cs;
      }
      Fr *= 0.0625f; Fi *= 0.0625f;     // ortho forward: 1/sqrt(256)
      size_t widx = (((size_t)kh * 9 + kw) * 256 + c) * 2;
      float wr = fmw[widx], wi = fmw[widx + 1];
      fs[ci][kh][kw][0] = Fr * wr - Fi * wi;
      fs[ci][kh][kw][1] = Fr * wi + Fi * wr;
    }
  }
  __syncthreads();

  { // inverse: iDFT over kh (h = t), irFFT over kw, residual add
    int h = t;
    float Gr[9], Gi[9];
#pragma unroll
    for (int kw = 0; kw < 9; ++kw){ Gr[kw] = 0.f; Gi[kw] = 0.f; }
    for (int kh = 0; kh < 16; ++kh){
      float a = w16 * (float)((kh * h) & 15);
      float cs = __cosf(a), sn = __sinf(a);
#pragma unroll
      for (int kw = 0; kw < 9; ++kw){
        float fr = fs[ci][kh][kw][0], fi = fs[ci][kh][kw][1];
        Gr[kw] += fr * cs - fi * sn;
        Gi[kw] += fr * sn + fi * cs;
      }
    }
#pragma unroll
    for (int w = 0; w < 16; ++w){
      float y = Gr[0] + ((w & 1) ? -Gr[8] : Gr[8]);
#pragma unroll
      for (int kw = 1; kw < 8; ++kw){
        float a = w16 * (float)((kw * w) & 15);
        y += 2.f * (Gr[kw] * __cosf(a) - Gi[kw] * __sinf(a));
      }
      y *= 0.0625f;                     // ortho inverse: 1/sqrt(256)
      size_t oi = ((size_t)(b * 192 + ph * 16 + h) * 192 + pw * 16 + w) * 256 + c;
      out[oi] = x1[oi] + y;
    }
  }
}

// ---------------------------------------------------------------------------
extern "C" void kernel_launch(void* const* d_in, const int* in_sizes, int n_in,
                              void* d_out, int out_size, void* d_ws, size_t ws_size,
                              hipStream_t stream){
  (void)in_sizes; (void)n_in; (void)out_size; (void)ws_size;
  const float* x     = (const float*)d_in[0];
  const float* n1g   = (const float*)d_in[1];
  const float* n1b   = (const float*)d_in[2];
  const float* qkvw  = (const float*)d_in[3];
  const float* bt0   = (const float*)d_in[4];
  const float* bt1   = (const float*)d_in[5];
  const float* bt2   = (const float*)d_in[6];
  const float* bt3   = (const float*)d_in[7];
  const float* projw = (const float*)d_in[8];
  const float* projb = (const float*)d_in[9];
  const float* fmw   = (const float*)d_in[10];
  const float* f1w   = (const float*)d_in[11];
  const float* f1b   = (const float*)d_in[12];
  const float* f2w   = (const float*)d_in[13];
  const float* f2b   = (const float*)d_in[14];
  const float* n2g   = (const float*)d_in[15];
  const float* n2b   = (const float*)d_in[16];

  const int M = 2 * 192 * 192;  // 73728 tokens

  char* p = (char*)d_ws;
  auto alloc = [&](size_t bytes) -> char* {
    char* r = p; p += (bytes + 255) & ~(size_t)255; return r;
  };
  __bf16* wQKV = (__bf16*)alloc((size_t)768 * 256 * 2);
  __bf16* wP   = (__bf16*)alloc((size_t)256 * 256 * 2);
  __bf16* wF1  = (__bf16*)alloc((size_t)1024 * 256 * 2);
  __bf16* wF2  = (__bf16*)alloc((size_t)256 * 1024 * 2);
  float* bias0 = (float*)alloc((size_t)2 * 256 * 256 * 4);
  float* bias1 = (float*)alloc((size_t)2 * 16 * 16 * 4);
  float* bias2 = (float*)alloc((size_t)2 * 64 * 64 * 4);
  float* bias3 = (float*)alloc((size_t)2 * 64 * 64 * 4);
  __bf16* xn   = (__bf16*)alloc((size_t)M * 256 * 2);
  __bf16* qkvb = (__bf16*)alloc((size_t)M * 768 * 2);
  __bf16* attb = (__bf16*)alloc((size_t)M * 256 * 2);
  float*  x1   = (float*)alloc((size_t)M * 256 * 4);
  __bf16* hn   = (__bf16*)alloc((size_t)M * 256 * 2);
  __bf16* gb   = (__bf16*)alloc((size_t)M * 1024 * 2);
  float*  h2   = (float*)alloc((size_t)M * 256 * 4);

  // weights -> bf16, bias tables -> dense matrices
  cvt_f32_bf16<<<(768 * 256 + 255) / 256, 256, 0, stream>>>(qkvw, wQKV, 768 * 256);
  cvt_f32_bf16<<<(256 * 256 + 255) / 256, 256, 0, stream>>>(projw, wP, 256 * 256);
  cvt_f32_bf16<<<(1024 * 256 + 255) / 256, 256, 0, stream>>>(f1w, wF1, 1024 * 256);
  cvt_f32_bf16<<<(256 * 1024 + 255) / 256, 256, 0, stream>>>(f2w, wF2, 256 * 1024);
  build_bias<<<(2 * 256 * 256 + 255) / 256, 256, 0, stream>>>(bt0, bias0, 16, 16);
  build_bias<<<(2 * 16 * 16 + 255) / 256, 256, 0, stream>>>(bt1, bias1, 4, 4);
  build_bias<<<(2 * 64 * 64 + 255) / 256, 256, 0, stream>>>(bt2, bias2, 4, 16);
  build_bias<<<(2 * 64 * 64 + 255) / 256, 256, 0, stream>>>(bt3, bias3, 16, 4);

  // LN1 -> bf16
  layernorm_bf16<<<M, 256, 0, stream>>>(x, n1g, n1b, xn);

  // qkv = LN(x) @ qkv_w^T  [M,768] bf16   (block tile 256x64)
  gemm_wmma<EPI_STORE_BF16><<<(M / 256) * (768 / 64), 256, 0, stream>>>(
      xn, wQKV, nullptr, nullptr, qkvb, M, 768, 256);

  // 4 shifted-window attentions (each exactly 9216 waves = 2304 blocks)
  win_attn_wmma<16, 16, 8, 8><<<2304, 128, 0, stream>>>(qkvb, bias0, attb, 0);
  win_attn_wmma< 4,  4, 2, 2><<<2304, 128, 0, stream>>>(qkvb, bias1, attb, 64);
  win_attn_wmma< 4, 16, 2, 8><<<2304, 128, 0, stream>>>(qkvb, bias2, attb, 128);
  win_attn_wmma<16,  4, 8, 2><<<2304, 128, 0, stream>>>(qkvb, bias3, attb, 192);

  // x1 = att @ proj_w^T + proj_b + x   (f32)
  gemm_wmma<EPI_BIAS_RES_F32><<<(M / 256) * (256 / 64), 256, 0, stream>>>(
      attb, wP, projb, x, x1, M, 256, 256);

  // LN2 -> bf16
  layernorm_bf16<<<M, 256, 0, stream>>>(x1, n2g, n2b, hn);

  // FFN: gelu(hn @ w1^T + b1) -> bf16 ; @ w2^T + b2 -> f32
  gemm_wmma<EPI_BIAS_GELU_BF16><<<(M / 256) * (1024 / 64), 256, 0, stream>>>(
      hn, wF1, f1b, nullptr, gb, M, 1024, 256);
  gemm_wmma<EPI_BIAS_F32><<<(M / 256) * (256 / 64), 256, 0, stream>>>(
      gb, wF2, f2b, nullptr, h2, M, 256, 1024);

  // out = x1 + freq_mod(h2)
  freq_mod_add<<<dim3(288, 16), 256, 0, stream>>>(h2, fmw, x1, (float*)d_out);
}
